// GRUNetBinaryEmbeding_88476326298277
// MI455X (gfx1250) — compile-verified, matrix operand order checked by reading, not run on hardware
//
#include <hip/hip_runtime.h>
#include <hip/hip_bf16.h>

typedef __bf16 v16bf __attribute__((ext_vector_type(16)));
typedef float  v8f   __attribute__((ext_vector_type(8)));

#define VOCAB 40
#define H     256
#define H3    768
#define B_    256
#define T_    512
#define MROWS 16      // batch rows per workgroup (= WMMA M)
#define NT    48      // N tiles (768/16)
#define KK    8       // K steps  (256/32)
#define TPW   6       // N tiles per wave (48 / 8 waves)

__device__ __forceinline__ unsigned short f2bf(float f) {
    union { float f; unsigned u; } v; v.f = f;
    unsigned r = v.u + 0x7FFFu + ((v.u >> 16) & 1u);   // RNE
    return (unsigned short)(r >> 16);
}
__device__ __forceinline__ float bf2f(unsigned short s) {
    union { unsigned u; float f; } v; v.u = ((unsigned)s) << 16;
    return v.f;
}

// ---------------------------------------------------------------------------
// P1: gi_table[v][j] = dot(embed[v,:], w_ih[j,:])   (40 x 768, tiny GEMM)
// ---------------------------------------------------------------------------
__global__ void gi_table_kernel(const float* __restrict__ embed,
                                const float* __restrict__ w_ih,
                                float* __restrict__ gi) {
    int id = blockIdx.x * blockDim.x + threadIdx.x;
    if (id >= VOCAB * H3) return;
    int v = id / H3, j = id - v * H3;
    const float* e = embed + v * H;
    const float* w = w_ih + j * H;
    float s = 0.f;
    #pragma unroll 8
    for (int k = 0; k < H; ++k) s = fmaf(e[k], w[k], s);
    gi[id] = s;
}

// ---------------------------------------------------------------------------
// P2: swizzle w_hh [768,256] f32 -> bf16 B-operand fragments.
// Fragment f = nt*8 + kk holds B tile (K = kk*32..+32, N = nt*16..+16),
// laid out per CDNA5 16-bit operand layout: lane l covers N = l&15,
// K-halves selected by l&16; 16 contiguous bf16 per lane (one 32B load).
// ---------------------------------------------------------------------------
__global__ void swizzle_whh_kernel(const float* __restrict__ w_hh,
                                   unsigned short* __restrict__ wB) {
    int d = blockIdx.x * blockDim.x + threadIdx.x;
    if (d >= H3 * H) return;
    int e    = d & 15;
    int lane = (d >> 4) & 31;
    int f    = d >> 9;
    int kk   = f & 7;
    int nt   = f >> 3;
    int n = nt * 16 + (lane & 15);
    int k = kk * 32 + ((lane & 16) ? 8 : 0) + ((e < 8) ? e : (8 + e));
    wB[d] = f2bf(w_hh[n * H + k]);     // B[k][n] = w_hh[n][k] (transposed use)
}

// ---------------------------------------------------------------------------
// Main persistent GRU kernel: 16 WGs x 256 threads (8 waves).
// Each WG owns 16 batch rows for the entire T=512 recurrence.
// ---------------------------------------------------------------------------
__global__ __launch_bounds__(256) void gru_persistent_kernel(
    const int*   __restrict__ x,
    const float* __restrict__ h0,
    const float* __restrict__ gi_table,
    const unsigned short* __restrict__ wB,
    const float* __restrict__ dec_w,
    const float* __restrict__ dec_b,
    float* __restrict__ d_out)
{
    __shared__ __align__(16) unsigned short hA[MROWS * H];   // 8 KB  bf16 hidden (A operand)
    __shared__ __align__(16) float gh[MROWS * H3];           // 48 KB gate pre-acts / decoder prods
    __shared__ float red[MROWS * 16];
    __shared__ int   sxv[MROWS];

    const int tid  = threadIdx.x;
    const int lane = tid & 31;
    const int wave = tid >> 5;
    const int b0   = blockIdx.x * MROWS;

    const float wdec = dec_w[tid];        // OUT==1, tid indexes the H dim
    const float bias = dec_b[0];

    // init hidden state from h0 (f32 -> bf16)
    for (int b = 0; b < MROWS; ++b)
        hA[b * H + tid] = f2bf(h0[(b0 + b) * H + tid]);
    __syncthreads();

    const int m     = lane & 15;
    const int khalf = (lane & 16) ? 8 : 0;
    const v8f vzero = {0.f, 0.f, 0.f, 0.f, 0.f, 0.f, 0.f, 0.f};

    for (int t = 0; t < T_; ++t) {
        if (tid < MROWS) sxv[tid] = x[(b0 + tid) * T_ + t];

        // ---- GEMM: gh[16 x 768] = hA[16 x 256] * w_hh^T, bf16 WMMA f32 acc
        v8f acc[TPW];
        #pragma unroll
        for (int i = 0; i < TPW; ++i) acc[i] = vzero;

        union Frag { v16bf v; uint4 q[2]; };
        #pragma unroll
        for (int kk = 0; kk < KK; ++kk) {
            Frag a;
            const uint4* ap = (const uint4*)&hA[m * H + kk * 32 + khalf];
            a.q[0] = ap[0];        // K = base .. base+7
            a.q[1] = ap[2];        // K = base+16 .. base+23
            #pragma unroll
            for (int i = 0; i < TPW; ++i) {
                const int nt = wave * TPW + i;
                Frag bf;
                const uint4* bp = (const uint4*)&wB[((nt * 8 + kk) << 9) + lane * 16];
                bf.q[0] = bp[0];
                bf.q[1] = bp[1];
                acc[i] = __builtin_amdgcn_wmma_f32_16x16x32_bf16(
                    false, a.v, false, bf.v, (short)0, acc[i], false, false);
            }
        }
        // scatter accumulators to gh (C/D layout: lane<16 -> M=j, else M=j+8)
        #pragma unroll
        for (int i = 0; i < TPW; ++i) {
            const int ncol = (wave * TPW + i) * 16 + m;
            #pragma unroll
            for (int j = 0; j < 8; ++j)
                gh[(j + khalf) * H3 + ncol] = acc[i][j];
        }
        __syncthreads();

        // ---- elementwise GRU cell; thread owns column c = tid across 16 rows
        const int c = tid;
        for (int b = 0; b < MROWS; ++b) {
            const float* gi = gi_table + sxv[b] * H3;
            float hr = gh[b * H3 + c];
            float hz = gh[b * H3 + c + H];
            float hn = gh[b * H3 + c + 2 * H];
            float r  = 1.f / (1.f + __expf(-(gi[c] + hr)));
            float z  = 1.f / (1.f + __expf(-(gi[c + H] + hz)));
            float n  = tanhf(gi[c + 2 * H] + r * hn);
            float ho = bf2f(hA[b * H + c]);
            float hnew = (1.f - z) * n + z * ho;
            hA[b * H + c] = f2bf(hnew);
            if (t == T_ - 1)
                d_out[B_ * T_ + (b0 + b) * H + c] = hnew;   // h_last [1,B,H]
            float act = hnew > 0.f ? hnew : 0.f;            // ReLU
            gh[b * H3 + c] = act * wdec;                    // decoder product
        }
        __syncthreads();

        // ---- decoder reduction: 256 -> 16 -> 1 per batch row
        {
            int b = tid >> 4, q = tid & 15;
            float s = 0.f;
            #pragma unroll
            for (int k = 0; k < 16; ++k) s += gh[b * H3 + q * 16 + k];
            red[b * 16 + q] = s;
        }
        __syncthreads();
        if (tid < MROWS) {
            float s = bias;
            #pragma unroll
            for (int q = 0; q < 16; ++q) s += red[tid * 16 + q];
            d_out[(b0 + tid) * T_ + t] = 1.f / (1.f + __expf(-s));
        }
        // next-iteration hazards covered by the three barriers above
    }
}

// ---------------------------------------------------------------------------
extern "C" void kernel_launch(void* const* d_in, const int* in_sizes, int n_in,
                              void* d_out, int out_size, void* d_ws, size_t ws_size,
                              hipStream_t stream) {
    (void)in_sizes; (void)n_in; (void)out_size; (void)ws_size;
    const int*   x     = (const int*)  d_in[0];
    const float* h0    = (const float*)d_in[1];
    const float* embed = (const float*)d_in[2];
    const float* w_ih  = (const float*)d_in[3];
    const float* w_hh  = (const float*)d_in[4];
    const float* dec_w = (const float*)d_in[5];
    const float* dec_b = (const float*)d_in[6];
    float* out = (float*)d_out;

    float*          gi_table = (float*)d_ws;
    unsigned short* wB = (unsigned short*)((char*)d_ws + (size_t)VOCAB * H3 * sizeof(float));

    hipLaunchKernelGGL(gi_table_kernel, dim3((VOCAB * H3 + 255) / 256), dim3(256), 0, stream,
                       embed, w_ih, gi_table);
    hipLaunchKernelGGL(swizzle_whh_kernel, dim3((H3 * H + 255) / 256), dim3(256), 0, stream,
                       w_hh, wB);
    hipLaunchKernelGGL(gru_persistent_kernel, dim3(B_ / MROWS), dim3(256), 0, stream,
                       x, h0, gi_table, wB, dec_w, dec_b, out);
}
